// LSTMDecoder_69123203661891
// MI455X (gfx1250) — compile-verified, matrix operand order checked by reading, not run on hardware
//
#include <hip/hip_runtime.h>
#include <math.h>

// ---------------------------------------------------------------------------
// LSTM decoder with dot attention, MI455X (gfx1250, wave32, WMMA).
//   - GEMMs (gates, output proj) use v_wmma_f32_16x16x32_bf16, one 16x16
//     C-tile per wave; weights pre-converted to bf16 (L2-resident, ~30MB).
//   - context optionally converted to bf16 once (halves the dominant HBM
//     traffic: 32 GB -> 16 GB over the 64 steps).
//   - attention score/align are batched GEMVs -> VALU + wave reductions.
// ---------------------------------------------------------------------------

typedef __bf16 v16bf  __attribute__((ext_vector_type(16)));
typedef float  v8f    __attribute__((ext_vector_type(8)));
typedef __bf16 bf16x2 __attribute__((ext_vector_type(2)));

constexpr int B = 64, T = 64, S = 1024, H = 1024, DIN = 1024;
constexpr int NG   = 4 * H;        // 4096 gate columns
constexpr int KCAT = DIN + 2 * H;  // 3072 (x | ctx_out | h)
constexpr int KOUT = 2 * H;        // 2048 (h | align)
constexpr int DVPS = 2 * H + DIN;  // 3072

// d_out layout (reference return order): ctx_outs, vps, attns, h, c
constexpr size_t OFF_CTXOUTS = 0;
constexpr size_t OFF_VPS     = (size_t)B * T * H;
constexpr size_t OFF_ATTNS   = OFF_VPS + (size_t)B * T * DVPS;
constexpr size_t OFF_HOUT    = OFF_ATTNS + (size_t)B * T * S;
constexpr size_t OFF_COUT    = OFF_HOUT + (size_t)B * H;

// ---------------------------------------------------------------------------
// helpers
// ---------------------------------------------------------------------------
__device__ __forceinline__ float2 ld2f(const float* p) {
    return *(const float2*)p;
}
__device__ __forceinline__ float2 ld2f(const __bf16* p) {
    bf16x2 v = *(const bf16x2*)p;
    float2 r; r.x = (float)v.x; r.y = (float)v.y;
    return r;
}

// ---------------------------------------------------------------------------
// one-time conversion kernels (per launch; deterministic)
// ---------------------------------------------------------------------------
__global__ void k_conv_wcat(const float* __restrict__ Wih,
                            const float* __restrict__ Whh,
                            __bf16* __restrict__ Wcat) {
    int idx = blockIdx.x * 256 + threadIdx.x;           // NG*KCAT
    if (idx >= NG * KCAT) return;
    int n = idx / KCAT, k = idx - n * KCAT;
    float v = (k < 2048) ? Wih[(size_t)n * 2048 + k]
                         : Whh[(size_t)n * 1024 + (k - 2048)];
    Wcat[idx] = (__bf16)v;
}

__global__ void k_conv_wout(const float* __restrict__ Wo,
                            __bf16* __restrict__ Wob) {
    int idx = blockIdx.x * 256 + threadIdx.x;           // H*KOUT
    if (idx >= H * KOUT) return;
    Wob[idx] = (__bf16)Wo[idx];
}

__global__ void k_conv_bcat(const float* __restrict__ bih,
                            const float* __restrict__ bhh,
                            float* __restrict__ bcat) {
    int idx = blockIdx.x * 256 + threadIdx.x;
    if (idx >= NG) return;
    bcat[idx] = bih[idx] + bhh[idx];
}

__global__ void k_conv_ctx(const float* __restrict__ ctxf,
                           __bf16* __restrict__ ctxb) {
    int idx = blockIdx.x * 256 + threadIdx.x;           // B*S*H = 2^26
    ctxb[idx] = (__bf16)ctxf[idx];
}

__global__ void k_init_state(const float* __restrict__ h0,
                             const float* __restrict__ c0,
                             float* __restrict__ h, float* __restrict__ c,
                             float* __restrict__ ctxo) {
    int idx = blockIdx.x * 256 + threadIdx.x;           // B*H
    h[idx] = h0[idx];
    c[idx] = c0[idx];
    ctxo[idx] = 0.0f;
}

// ---------------------------------------------------------------------------
// gates GEMM: gates[64, 4096] = [x_t | ctx_out | h] @ Wcat^T   (K = 3072)
// One wave32 per 16x16 tile; K stepped 32 at a time via bf16 WMMA.
// ---------------------------------------------------------------------------
__global__ void k_gates_gemm(const float* __restrict__ x, int t,
                             const float* __restrict__ ctxo,
                             const float* __restrict__ h,
                             const __bf16* __restrict__ Wcat,
                             float* __restrict__ gates) {
    const int lane = threadIdx.x & 31;
    const int wave = threadIdx.x >> 5;
    const int n0 = (blockIdx.x * 4 + wave) * 16;        // gate column tile
    const int m0 = blockIdx.y * 16;                     // batch row tile
    const int half = lane >> 4, l16 = lane & 15;
    const int m = m0 + l16;                             // A row (batch)
    const int n = n0 + l16;                             // B row (gate col)

    const float* abase[3] = { x    + ((size_t)m * T + t) * DIN,
                              ctxo + (size_t)m * H,
                              h    + (size_t)m * H };
    const __bf16* brow = Wcat + (size_t)n * KCAT;

    v8f acc = {};
#pragma unroll
    for (int reg = 0; reg < 3; ++reg) {
        const float* ap0 = abase[reg];
        for (int kk = 0; kk < 1024; kk += 32) {
            v16bf a, bq;
#pragma unroll
            for (int j = 0; j < 8; ++j) {
                // ISA 16-bit A/B layout: VGPR j holds K = {0|16} + 2*(j&3) + 8*half
                int kp = ((j & 4) << 2) + ((j & 3) << 1) + (half << 3);
                float2 av = *(const float2*)(ap0 + kk + kp);
                a[2 * j]     = (__bf16)av.x;
                a[2 * j + 1] = (__bf16)av.y;
                bf16x2 bv = *(const bf16x2*)(brow + reg * 1024 + kk + kp);
                bq[2 * j]     = bv.x;
                bq[2 * j + 1] = bv.y;
            }
            acc = __builtin_amdgcn_wmma_f32_16x16x32_bf16(
                false, a, false, bq, (short)0, acc, false, false);
        }
    }
#pragma unroll
    for (int r = 0; r < 8; ++r) {
        int row = m0 + r + half * 8;                    // C/D layout
        gates[(size_t)row * NG + n0 + l16] = acc[r];
    }
}

// ---------------------------------------------------------------------------
// LSTM cell elementwise (gate order i, f, g, o as in torch LSTMCell)
// ---------------------------------------------------------------------------
__global__ void k_lstm_cell(const float* __restrict__ gates,
                            const float* __restrict__ bcat,
                            float* __restrict__ c, float* __restrict__ h) {
    int idx = blockIdx.x * 256 + threadIdx.x;           // B*H
    int b = idx >> 10, d = idx & (H - 1);
    const float* g = gates + (size_t)b * NG;
    float gi = g[d]          + bcat[d];
    float gf = g[1024 + d]   + bcat[1024 + d];
    float gg = g[2048 + d]   + bcat[2048 + d];
    float go = g[3072 + d]   + bcat[3072 + d];
    float si = 1.0f / (1.0f + expf(-gi));
    float sf = 1.0f / (1.0f + expf(-gf));
    float so = 1.0f / (1.0f + expf(-go));
    float cn = sf * c[idx] + si * tanhf(gg);
    float hn = so * tanhf(cn);
    c[idx] = cn;
    h[idx] = hn;
}

// ---------------------------------------------------------------------------
// attention scores: scores[b,s] = h[b,:] . context[b,s,:]  (masked)
// one wave per (b,s); coalesced dword loads + wave shuffle reduction
// ---------------------------------------------------------------------------
template <typename CT>
__global__ void k_scores(const float* __restrict__ h,
                         const CT* __restrict__ ctx,
                         const unsigned char* __restrict__ mask,
                         float* __restrict__ scores) {
    int p = blockIdx.x * 8 + (threadIdx.x >> 5);        // (b,s) pair id
    int lane = threadIdx.x & 31;
    int b = p >> 10, s = p & (S - 1);
    const CT* crow = ctx + ((size_t)b * S + s) * H;
    const float* hrow = h + (size_t)b * H;
    float acc = 0.0f;
    for (int k = lane * 2; k < H; k += 64) {
        float2 cv = ld2f(crow + k);
        float2 hv = *(const float2*)(hrow + k);
        acc += cv.x * hv.x + cv.y * hv.y;
    }
    for (int o = 16; o > 0; o >>= 1) acc += __shfl_xor(acc, o, 32);
    if (lane == 0) {
        size_t i = (size_t)b * S + s;
        scores[i] = mask[i] ? acc : -INFINITY;
    }
}

// ---------------------------------------------------------------------------
// softmax over S per batch; writes attn straight into d_out attns region
// ---------------------------------------------------------------------------
__global__ void k_softmax(const float* __restrict__ scores,
                          float* __restrict__ attnAll, int t) {
    int b = blockIdx.x, tid = threadIdx.x;
    __shared__ float red[256];
    const float* srow = scores + (size_t)b * S;
    float v0 = srow[tid], v1 = srow[tid + 256], v2 = srow[tid + 512], v3 = srow[tid + 768];
    float mx = fmaxf(fmaxf(v0, v1), fmaxf(v2, v3));
    red[tid] = mx; __syncthreads();
    for (int w = 128; w > 0; w >>= 1) {
        if (tid < w) red[tid] = fmaxf(red[tid], red[tid + w]);
        __syncthreads();
    }
    mx = red[0]; __syncthreads();
    float e0 = expf(v0 - mx), e1 = expf(v1 - mx), e2 = expf(v2 - mx), e3 = expf(v3 - mx);
    red[tid] = e0 + e1 + e2 + e3; __syncthreads();
    for (int w = 128; w > 0; w >>= 1) {
        if (tid < w) red[tid] += red[tid + w];
        __syncthreads();
    }
    float inv = 1.0f / red[0];
    float* arow = attnAll + ((size_t)b * T + t) * S;
    arow[tid]       = e0 * inv;
    arow[tid + 256] = e1 * inv;
    arow[tid + 512] = e2 * inv;
    arow[tid + 768] = e3 * inv;
}

// ---------------------------------------------------------------------------
// align[b,d] = sum_s attn[b,s] * context[b,s,d]; attn chunk staged in LDS,
// each thread owns 4 contiguous d columns (coalesced across the block)
// ---------------------------------------------------------------------------
template <typename CT>
__global__ void k_align(const float* __restrict__ attnAll, int t,
                        const CT* __restrict__ ctx,
                        float* __restrict__ alignv) {
    int b = blockIdx.x, tid = threadIdx.x;
    __shared__ float sa[256];
    int d0 = tid * 4;
    float a0 = 0, a1 = 0, a2 = 0, a3 = 0;
    const float* arow = attnAll + ((size_t)b * T + t) * S;
    for (int s0 = 0; s0 < S; s0 += 256) {
        sa[tid] = arow[s0 + tid];
        __syncthreads();
        for (int s2 = 0; s2 < 256; ++s2) {
            float a = sa[s2];
            const CT* row = ctx + ((size_t)b * S + s0 + s2) * H + d0;
            float2 p0 = ld2f(row);
            float2 p1 = ld2f(row + 2);
            a0 += a * p0.x; a1 += a * p0.y; a2 += a * p1.x; a3 += a * p1.y;
        }
        __syncthreads();
    }
    float* dst = alignv + (size_t)b * H + d0;
    dst[0] = a0; dst[1] = a1; dst[2] = a2; dst[3] = a3;
}

// ---------------------------------------------------------------------------
// output projection: ctx_new = tanh([h | align] @ W_out^T)   (K = 2048)
// epilogue writes ctx_out state + ctx_outs output + vps middle slice
// ---------------------------------------------------------------------------
__global__ void k_out_gemm(const float* __restrict__ h,
                           const float* __restrict__ alignv,
                           const __bf16* __restrict__ Wout, int t,
                           float* __restrict__ ctxo,
                           float* __restrict__ ctxOuts,
                           float* __restrict__ vps) {
    const int lane = threadIdx.x & 31;
    const int wave = threadIdx.x >> 5;
    const int n0 = (blockIdx.x * 4 + wave) * 16;
    const int m0 = blockIdx.y * 16;
    const int half = lane >> 4, l16 = lane & 15;
    const int m = m0 + l16, n = n0 + l16;

    const float* abase[2] = { h + (size_t)m * H, alignv + (size_t)m * H };
    const __bf16* brow = Wout + (size_t)n * KOUT;

    v8f acc = {};
#pragma unroll
    for (int reg = 0; reg < 2; ++reg) {
        const float* ap0 = abase[reg];
        for (int kk = 0; kk < 1024; kk += 32) {
            v16bf a, bq;
#pragma unroll
            for (int j = 0; j < 8; ++j) {
                int kp = ((j & 4) << 2) + ((j & 3) << 1) + (half << 3);
                float2 av = *(const float2*)(ap0 + kk + kp);
                a[2 * j]     = (__bf16)av.x;
                a[2 * j + 1] = (__bf16)av.y;
                bf16x2 bv = *(const bf16x2*)(brow + reg * 1024 + kk + kp);
                bq[2 * j]     = bv.x;
                bq[2 * j + 1] = bv.y;
            }
            acc = __builtin_amdgcn_wmma_f32_16x16x32_bf16(
                false, a, false, bq, (short)0, acc, false, false);
        }
    }
#pragma unroll
    for (int r = 0; r < 8; ++r) {
        int row = m0 + r + half * 8;
        int col = n0 + l16;
        float v = tanhf(acc[r]);
        ctxo[(size_t)row * H + col] = v;
        ctxOuts[((size_t)row * T + t) * H + col] = v;
        vps[((size_t)row * T + t) * DVPS + H + col] = v;
    }
}

// vps[:, t, 0:H] = h_new ; vps[:, t, 2H:3H] = x_t
__global__ void k_vps_write(const float* __restrict__ h,
                            const float* __restrict__ x, int t,
                            float* __restrict__ vps) {
    int idx = blockIdx.x * 256 + threadIdx.x;           // B*H
    int b = idx >> 10, d = idx & (H - 1);
    float* row = vps + ((size_t)b * T + t) * DVPS;
    row[d]          = h[idx];
    row[2 * H + d]  = x[((size_t)b * T + t) * DIN + d];
}

__global__ void k_finalize(const float* __restrict__ h,
                           const float* __restrict__ c,
                           float* __restrict__ hOut, float* __restrict__ cOut) {
    int idx = blockIdx.x * 256 + threadIdx.x;           // B*H
    hOut[idx] = h[idx];
    cOut[idx] = c[idx];
}

// ---------------------------------------------------------------------------
// host driver
// ---------------------------------------------------------------------------
extern "C" void kernel_launch(void* const* d_in, const int* in_sizes, int n_in,
                              void* d_out, int out_size, void* d_ws, size_t ws_size,
                              hipStream_t stream) {
    (void)in_sizes; (void)n_in; (void)out_size;

    const float* x      = (const float*)d_in[0];
    const float* ctxf   = (const float*)d_in[1];
    const unsigned char* cmask = (const unsigned char*)d_in[2];
    const float* W_ih   = (const float*)d_in[3];
    const float* b_ih   = (const float*)d_in[4];
    const float* W_hh   = (const float*)d_in[5];
    const float* b_hh   = (const float*)d_in[6];
    const float* W_out  = (const float*)d_in[7];
    const float* h0     = (const float*)d_in[8];
    const float* c0     = (const float*)d_in[9];
    float* out = (float*)d_out;

    // workspace carve-up (256B aligned slices)
    char* wsB = (char*)d_ws;
    size_t off = 0;
    auto take = [&](size_t bytes) -> void* {
        void* p = wsB + off;
        off = (off + bytes + 255) & ~(size_t)255;
        return p;
    };
    __bf16* Wcat   = (__bf16*)take((size_t)NG * KCAT * 2);   // 25.2 MB
    __bf16* Woutb  = (__bf16*)take((size_t)H * KOUT * 2);    //  4.2 MB
    float*  bcat   = (float*) take((size_t)NG * 4);
    float*  h      = (float*) take((size_t)B * H * 4);
    float*  c      = (float*) take((size_t)B * H * 4);
    float*  ctxo   = (float*) take((size_t)B * H * 4);
    float*  alignv = (float*) take((size_t)B * H * 4);
    float*  gates  = (float*) take((size_t)B * NG * 4);      //  1 MB
    float*  scores = (float*) take((size_t)B * S * 4);
    __bf16* ctxbf  = (__bf16*)take((size_t)B * S * H * 2);   // 128 MB (optional)
    const bool useBf16Ctx = (ws_size >= off);

    // --- one-time prep (each call; deterministic) ---
    k_conv_wcat<<<(NG * KCAT + 255) / 256, 256, 0, stream>>>(W_ih, W_hh, Wcat);
    k_conv_wout<<<(H * KOUT + 255) / 256, 256, 0, stream>>>(W_out, Woutb);
    k_conv_bcat<<<(NG + 255) / 256, 256, 0, stream>>>(b_ih, b_hh, bcat);
    k_init_state<<<(B * H) / 256, 256, 0, stream>>>(h0, c0, h, c, ctxo);
    if (useBf16Ctx) {
        k_conv_ctx<<<(unsigned)(((size_t)B * S * H) / 256), 256, 0, stream>>>(ctxf, ctxbf);
    }

    // --- sequential decode over T ---
    for (int t = 0; t < T; ++t) {
        k_gates_gemm<<<dim3(NG / 64, B / 16), 128, 0, stream>>>(x, t, ctxo, h, Wcat, gates);
        k_lstm_cell<<<(B * H) / 256, 256, 0, stream>>>(gates, bcat, c, h);
        if (useBf16Ctx)
            k_scores<__bf16><<<(B * S) / 8, 256, 0, stream>>>(h, ctxbf, cmask, scores);
        else
            k_scores<float><<<(B * S) / 8, 256, 0, stream>>>(h, ctxf, cmask, scores);
        k_softmax<<<B, 256, 0, stream>>>(scores, out + OFF_ATTNS, t);
        if (useBf16Ctx)
            k_align<__bf16><<<B, 256, 0, stream>>>(out + OFF_ATTNS, t, ctxbf, alignv);
        else
            k_align<float><<<B, 256, 0, stream>>>(out + OFF_ATTNS, t, ctxf, alignv);
        k_out_gemm<<<dim3(H / 64, B / 16), 128, 0, stream>>>(h, alignv, Woutb, t, ctxo,
                                                             out + OFF_CTXOUTS, out + OFF_VPS);
        k_vps_write<<<(B * H) / 256, 256, 0, stream>>>(h, x, t, out + OFF_VPS);
    }

    k_finalize<<<(B * H) / 256, 256, 0, stream>>>(h, c, out + OFF_HOUT, out + OFF_COUT);
}
